// Mamba2Counter_670014898336
// MI455X (gfx1250) — matrix operand order repaired and checked by reading, not compiled, not run on hardware
//
#include <hip/hip_runtime.h>
#include <hip/hip_bf16.h>
#include <math.h>

// ---------------- CDNA5 / problem constants ----------------
typedef _Float16 v16h __attribute__((ext_vector_type(16)));
typedef float    v8f  __attribute__((ext_vector_type(8)));
typedef int      v4i  __attribute__((vector_size(16)));   // matches async-LDS builtin param

#define D_MODEL 256
#define D_INNER 512
#define D_STATE 16
#define NLAYERS 4
#define SEQ     900
#define BATCH   64
#define MROWS   (BATCH * SEQ)   // 57600 rows, divisible by 64
#define LN_EPS  1e-5f

#ifndef __has_builtin
#define __has_builtin(x) 0
#endif
#if __has_builtin(__builtin_amdgcn_global_load_async_to_lds_b128)
#define HAVE_ASYNC_LDS 1
#else
#define HAVE_ASYNC_LDS 0
#endif
#if __has_builtin(__builtin_amdgcn_s_wait_asynccnt)
#define HAVE_WAIT_ASYNC 1
#else
#define HAVE_WAIT_ASYNC 0
#endif

__device__ __forceinline__ float silu_f(float v) { return v / (1.f + __expf(-v)); }

union FragH { v16h v; uint4 u[2]; };

// 16-byte global -> LDS copy. Async (ASYNCcnt-tracked, no VGPR staging) when
// the gfx1250 builtin is available; plain b128 load/store otherwise.
// Builtin signature (from hipcc diagnostic): arg0 = v4i* in AS(1) (printed
// as "__device__"), non-const; 4 args total (arity confirmed by clang
// checking arity before types).
__device__ __forceinline__ void cp16_lds(_Float16* lds, const _Float16* g)
{
#if HAVE_ASYNC_LDS
    __builtin_amdgcn_global_load_async_to_lds_b128(
        (__attribute__((address_space(1))) v4i*)g,
        (__attribute__((address_space(3))) v4i*)lds, 0, 0);
#else
    *(uint4*)lds = *(const uint4*)g;
#endif
}

// Wait until this wave's outstanding async copies <= n (n in {0,2,4}).
__device__ __forceinline__ void wait_async_le(int n)
{
#if HAVE_ASYNC_LDS
#if HAVE_WAIT_ASYNC
    if (n >= 4)      __builtin_amdgcn_s_wait_asynccnt(4);
    else if (n >= 2) __builtin_amdgcn_s_wait_asynccnt(2);
    else             __builtin_amdgcn_s_wait_asynccnt(0);
#else
    if (n >= 4)      asm volatile("s_wait_asynccnt 0x4" ::: "memory");
    else if (n >= 2) asm volatile("s_wait_asynccnt 0x2" ::: "memory");
    else             asm volatile("s_wait_asynccnt 0x0" ::: "memory");
#endif
#else
    (void)n;
#endif
}

// =====================================================================
// Tiled WMMA GEMM, double-buffered with async global->LDS staging:
//   C[M x N] = A[M x K] * Bt[N x K]^T + bias (+ optional f32 residual)
// AHALF: A is f16 (async b128 copy to LDS) else f32 (VALU convert while staging)
// CHALF: C stored f16 (activation stream)  else f32 (+residual)
// 128 threads = 4 waves; block tile 64x64; K-step 32 = one
// v_wmma_f32_16x16x32_f16 per fragment pair; f32 accumulation.
// Pipeline: issue tile t+1 copies -> s_wait_asynccnt <= per_tile (retires
// tile t, copies are in-order) -> barrier -> WMMA on tile t -> barrier.
// =====================================================================
template <bool AHALF, bool CHALF>
__global__ __launch_bounds__(128)
void gemm_wmma(const void* __restrict__ Aptr, int lda,
               const _Float16* __restrict__ Bt,             // [N][K] transposed weights
               const float* __restrict__ bias,              // [N]
               const float* __restrict__ Cadd, int ldadd,   // f32 residual (nullable, !CHALF)
               void* __restrict__ Cptr, int ldc,
               int N, int K)
{
    __shared__ __align__(16) _Float16 As[2][64 * 48];   // 64 rows x 32 K, stride 48 halfs (96B)
    __shared__ __align__(16) _Float16 Bs[2][64 * 48];

    const int tid  = threadIdx.x;
    const int lane = tid & 31;
    const int wave = tid >> 5;
    const int bm   = blockIdx.x * 64;
    const int bn   = blockIdx.y * 64;
    const int wm   = (wave >> 1) * 32;
    const int wn   = (wave & 1) * 32;

    const int lrow = tid >> 1;          // 0..63 (wave w stages rows w*16..w*16+15)
    const int lcb  = (tid & 1) * 16;    // 0 or 16

    // wave-uniform: does this wave's B-stage row range fall inside N?
    // (N is always a multiple of 16, so the whole wave agrees)
    const bool bwave_ok = (bn + (lrow & ~15)) < N;
    const int  per_tile = (AHALF ? 2 : 0) + (bwave_ok ? 2 : 0);  // async ops/tile/wave

    auto stageA = [&](int k0, int buf) {
        const int m = bm + lrow;
        _Float16* dst = &As[buf][lrow * 48 + lcb];
        if (AHALF) {
            const _Float16* ap = (const _Float16*)Aptr + (size_t)m * lda + k0 + lcb;
            cp16_lds(dst, ap);
            cp16_lds(dst + 8, ap + 8);
        } else {
            const float* ap = (const float*)Aptr + (size_t)m * lda + k0 + lcb;
#pragma unroll
            for (int i = 0; i < 16; ++i) dst[i] = (_Float16)ap[i];  // v_cvt_pk_f16_f32
        }
    };
    auto stageB = [&](int k0, int buf) {
        const int n = bn + lrow;
        _Float16* dst = &Bs[buf][lrow * 48 + lcb];
        if (n < N) {
            const _Float16* bp = Bt + (size_t)n * K + k0 + lcb;
            cp16_lds(dst, bp);
            cp16_lds(dst + 8, bp + 8);
        } else {
            const uint4 z = {0u, 0u, 0u, 0u};
            *(uint4*)dst = z;
            *(uint4*)(dst + 8) = z;
        }
    };

    v8f acc[2][2] = {};
    const int nk = K >> 5;

    stageA(0, 0);
    stageB(0, 0);

    for (int t = 0; t < nk; ++t) {
        const int cur = t & 1;
        if (t + 1 < nk) {             // prefetch next tile into the other buffer
            stageA((t + 1) << 5, cur ^ 1);
            stageB((t + 1) << 5, cur ^ 1);
        }
        wait_async_le((t + 1 < nk) ? per_tile : 0);   // retire tile t's copies
        __syncthreads();

        // ---- fragment loads per documented wave32 layouts ----
        // A (16x32 f16): lanes 0-15: M=lane,    K = {0..7, 16..23}
        //                lanes 16-31: M=lane-16, K = {8..15, 24..31}
        // B (32x16 f16): lanes 0-15: N=lane,    K = {0..7, 16..23}
        //                lanes 16-31: N=lane-16, K = {8..15, 24..31}
        const int r    = lane & 15;
        const int ksel = lane >> 4;
        FragH af[2], bf[2];
#pragma unroll
        for (int fm = 0; fm < 2; ++fm) {
            const int base = (wm + fm * 16 + r) * 48 + ksel * 8;
            af[fm].u[0] = *(const uint4*)&As[cur][base];
            af[fm].u[1] = *(const uint4*)&As[cur][base + 16];
        }
#pragma unroll
        for (int fn = 0; fn < 2; ++fn) {
            const int base = (wn + fn * 16 + r) * 48 + ksel * 8;
            bf[fn].u[0] = *(const uint4*)&Bs[cur][base];
            bf[fn].u[1] = *(const uint4*)&Bs[cur][base + 16];
        }
#pragma unroll
        for (int fm = 0; fm < 2; ++fm)
#pragma unroll
            for (int fn = 0; fn < 2; ++fn)
                acc[fm][fn] = __builtin_amdgcn_wmma_f32_16x16x32_f16(
                    false, af[fm].v, false, bf[fn].v, (short)0, acc[fm][fn], false, false);
        __syncthreads();   // all waves done reading buf `cur` before it is re-staged
    }

    // ---- store D (VGPR r: lanes 0-15 -> M=r, lanes 16-31 -> M=r+8) ----
    const int col = lane & 15;
    const int rb  = (lane >> 4) * 8;
#pragma unroll
    for (int fm = 0; fm < 2; ++fm)
#pragma unroll
        for (int fn = 0; fn < 2; ++fn)
#pragma unroll
            for (int rr = 0; rr < 8; ++rr) {
                const int gm = bm + wm + fm * 16 + rb + rr;
                const int gn = bn + wn + fn * 16 + col;
                if (gn < N) {
                    float v = acc[fm][fn][rr] + bias[gn];
                    if (CHALF) {
                        ((_Float16*)Cptr)[(size_t)gm * ldc + gn] = (_Float16)v;
                    } else {
                        if (Cadd) v += Cadd[(size_t)gm * ldadd + gn];
                        ((float*)Cptr)[(size_t)gm * ldc + gn] = v;
                    }
                }
            }
}

// =====================================================================
// Embedding: x[m][d] = color_embed[grid[m]][d] + pos_embed[m % SEQ][d]
// =====================================================================
__global__ void embed_kernel(const int* __restrict__ grid, const float* __restrict__ ce,
                             const float* __restrict__ pe, float* __restrict__ x)
{
    const int idx = blockIdx.x * blockDim.x + threadIdx.x;
    if (idx >= MROWS * D_MODEL) return;
    const int m = idx >> 8, d = idx & 255;
    const int s = m % SEQ;
    const int g = grid[m];
    x[idx] = ce[g * 256 + d] + pe[s * 256 + d];
}

// =====================================================================
// Weight transpose + f32->f16 conversion (once per call, tiny)
// =====================================================================
__global__ void convert_weights_kernel(const float* __restrict__ in_w,
                                       const float* __restrict__ out_w,
                                       const float* __restrict__ xp_w,
                                       _Float16* __restrict__ in_t,
                                       _Float16* __restrict__ out_t,
                                       _Float16* __restrict__ xp_t)
{
    const int idx  = blockIdx.x * blockDim.x + threadIdx.x;
    const int NIN  = NLAYERS * 1024 * 256;
    const int NOUT = NLAYERS * 256 * 512;
    const int NXP  = NLAYERS * 32 * 512;
    if (idx < NIN) {
        const int l = idx / (1024 * 256), r = idx % (1024 * 256);
        const int n = r / 256, k = r % 256;
        in_t[idx] = (_Float16)in_w[(size_t)l * 256 * 1024 + (size_t)k * 1024 + n];
    } else if (idx < NIN + NOUT) {
        const int j = idx - NIN;
        const int l = j / (256 * 512), r = j % (256 * 512);
        const int n = r / 512, k = r % 512;
        out_t[j] = (_Float16)out_w[(size_t)l * 512 * 256 + (size_t)k * 256 + n];
    } else if (idx < NIN + NOUT + NXP) {
        const int j = idx - NIN - NOUT;
        const int l = j / (32 * 512), r = j % (32 * 512);
        const int n = r / 512, k = r % 512;
        xp_t[j] = (_Float16)xp_w[(size_t)l * 512 * 32 + (size_t)k * 32 + n];
    }
}

// =====================================================================
// Causal depthwise conv (width 4) + SiLU, f16 in / f16 out
// xs = xrh[:, :512] of the f16 in-proj output
// =====================================================================
__global__ void conv_silu_kernel(const _Float16* __restrict__ xrh,
                                 const float* __restrict__ cw,  // [512][4]
                                 const float* __restrict__ cb,
                                 _Float16* __restrict__ out)
{
    const int idx = blockIdx.x * blockDim.x + threadIdx.x;
    if (idx >= MROWS * D_INNER) return;
    const int m = idx >> 9, d = idx & 511;
    const int s = m % SEQ;
    const int base = m - s;
    float acc = cb[d];
#pragma unroll
    for (int k = 0; k < 4; ++k) {
        const int sp = s - 3 + k;
        if (sp >= 0) acc += (float)xrh[(size_t)(base + sp) * 1024 + d] * cw[d * 4 + k];
    }
    out[idx] = (_Float16)silu_f(acc);
}

// =====================================================================
// Selective scan + fused output gate.
// Block = 16 channels x 16 states (256 thr, 8 waves); fp32 recurrence.
// 16-lane butterfly reduction stays inside one wave32 half.
// Writeback (lane n==0): y_gated = (sum_n h*B + D*x) * silu(res), f16,
// written in place over the xs buffer (read-before-write within the wave).
// =====================================================================
__global__ __launch_bounds__(256)
void scan_kernel(const float* __restrict__ ssm, const float* __restrict__ A_log,
                 const float* __restrict__ Dp, const _Float16* __restrict__ xrh,
                 _Float16* __restrict__ xy)
{
    const int t      = threadIdx.x;
    const int b      = blockIdx.x >> 5;
    const int dchunk = blockIdx.x & 31;
    const int d      = dchunk * 16 + (t >> 4);
    const int n      = t & 15;
    const float Ac   = -__expf(A_log[d * 16 + n]);
    const float Dv   = Dp[d];
    float h = 0.f;
    const size_t mbase = (size_t)b * SEQ;
    for (int s = 0; s < SEQ; ++s) {
        const size_t m = mbase + s;
        float dn = ssm[m * 32 + n];
        dn = (dn > 20.f) ? dn : log1pf(__expf(dn));           // softplus
        const float bn = ssm[m * 32 + 16 + n];
        const float xv = (float)xy[m * 512 + d];
        h = h * __expf(Ac * dn) + xv * bn;
        float p = h * bn;
        p += __shfl_xor(p, 8, 32);
        p += __shfl_xor(p, 4, 32);
        p += __shfl_xor(p, 2, 32);
        p += __shfl_xor(p, 1, 32);
        if (n == 0) {
            const float res = (float)xrh[m * 1024 + 512 + d];
            xy[m * 512 + d] = (_Float16)((p + Dv * xv) * silu_f(res));
        }
    }
}

// =====================================================================
// LayerNorm over d=256, in place, one block per row (fp32 stream)
// =====================================================================
__global__ __launch_bounds__(256)
void layernorm_kernel(float* __restrict__ x, const float* __restrict__ g,
                      const float* __restrict__ b)
{
    __shared__ float red[256];
    const size_t m = blockIdx.x;
    const int t = threadIdx.x;
    const float v = x[m * 256 + t];
    red[t] = v;
    __syncthreads();
    for (int off = 128; off > 0; off >>= 1) { if (t < off) red[t] += red[t + off]; __syncthreads(); }
    const float mu = red[0] * (1.f / 256.f);
    __syncthreads();
    const float dv = v - mu;
    red[t] = dv * dv;
    __syncthreads();
    for (int off = 128; off > 0; off >>= 1) { if (t < off) red[t] += red[t + off]; __syncthreads(); }
    const float var = red[0] * (1.f / 256.f);
    x[m * 256 + t] = dv * rsqrtf(var + LN_EPS) * g[t] + b[t];
}

// =====================================================================
// Heads (tiny): layer-1 of both MLPs, then layer-2 + ReLU into d_out
// =====================================================================
__global__ __launch_bounds__(128)
void heads1_kernel(const float* __restrict__ x,
                   const float* __restrict__ c1w, const float* __restrict__ c1b,
                   const float* __restrict__ l1w, const float* __restrict__ l1b,
                   float* __restrict__ hbuf)
{
    __shared__ float fin[256];
    const int b = blockIdx.x, t = threadIdx.x;
    const size_t row = ((size_t)b * SEQ + (SEQ - 1)) * 256;
    fin[t] = x[row + t];
    fin[t + 128] = x[row + t + 128];
    __syncthreads();
    float a0 = c1b[t], a1 = l1b[t];
    for (int k = 0; k < 256; ++k) {
        const float f = fin[k];
        a0 += f * c1w[k * 128 + t];
        a1 += f * l1w[k * 128 + t];
    }
    hbuf[b * 256 + t]       = fmaxf(a0, 0.f);
    hbuf[b * 256 + 128 + t] = fmaxf(a1, 0.f);
}

__global__ __launch_bounds__(128)
void heads2_kernel(const float* __restrict__ hbuf,
                   const float* __restrict__ c2w, const float* __restrict__ c2b,
                   const float* __restrict__ l2w, const float* __restrict__ l2b,
                   float* __restrict__ out)
{
    __shared__ float red[128];
    const int b = blockIdx.x, t = threadIdx.x;
    const float t1 = hbuf[b * 256 + t];
    const float c1 = hbuf[b * 256 + 128 + t];
    red[t] = t1 * c2w[t];
    __syncthreads();
    for (int off = 64; off > 0; off >>= 1) { if (t < off) red[t] += red[t + off]; __syncthreads(); }
    if (t == 0) out[b] = fmaxf(red[0] + c2b[0], 0.f);
    __syncthreads();
    for (int c = 0; c < 10; ++c) {
        red[t] = c1 * l2w[t * 10 + c];
        __syncthreads();
        for (int off = 64; off > 0; off >>= 1) { if (t < off) red[t] += red[t + off]; __syncthreads(); }
        if (t == 0) out[64 + b * 10 + c] = fmaxf(red[0] + l2b[c], 0.f);
        __syncthreads();
    }
}

// =====================================================================
// Launch
// =====================================================================
extern "C" void kernel_launch(void* const* d_in, const int* in_sizes, int n_in,
                              void* d_out, int out_size, void* d_ws, size_t ws_size,
                              hipStream_t stream)
{
    const int*   grid   = (const int*)  d_in[0];
    const float* ce     = (const float*)d_in[1];
    const float* pe     = (const float*)d_in[2];
    const float* in_w   = (const float*)d_in[3];
    const float* in_b   = (const float*)d_in[4];
    const float* conv_w = (const float*)d_in[5];
    const float* conv_b = (const float*)d_in[6];
    const float* xp_w   = (const float*)d_in[7];
    const float* xp_b   = (const float*)d_in[8];
    const float* A_log  = (const float*)d_in[9];
    const float* Dp     = (const float*)d_in[10];
    const float* out_w  = (const float*)d_in[11];
    const float* out_b  = (const float*)d_in[12];
    const float* ln_g   = (const float*)d_in[13];
    const float* ln_b   = (const float*)d_in[14];
    const float* c1w    = (const float*)d_in[15];
    const float* c1b    = (const float*)d_in[16];
    const float* c2w    = (const float*)d_in[17];
    const float* c2b    = (const float*)d_in[18];
    const float* l1w    = (const float*)d_in[19];
    const float* l1b    = (const float*)d_in[20];
    const float* l2w    = (const float*)d_in[21];
    const float* l2b    = (const float*)d_in[22];
    float* outp = (float*)d_out;
    (void)in_sizes; (void)n_in; (void)out_size; (void)ws_size;

    // workspace layout (~248 MB)
    char* ws = (char*)d_ws;
    size_t off = 0;
    auto wsalloc = [&](size_t bytes) { void* p = ws + off; off += (bytes + 255) & ~(size_t)255; return p; };
    float*    X    = (float*)   wsalloc((size_t)MROWS * 256  * 4);   // fp32 residual stream
    _Float16* XRH  = (_Float16*)wsalloc((size_t)MROWS * 1024 * 2);   // f16 in-proj out (xs | res)
    _Float16* XSCH = (_Float16*)wsalloc((size_t)MROWS * 512  * 2);   // f16 conv out; scan writes gated y in place
    float*    SSM  = (float*)   wsalloc((size_t)MROWS * 32   * 4);   // fp32 (delta | B) per (b,s)
    _Float16* WIN  = (_Float16*)wsalloc((size_t)NLAYERS * 1024 * 256 * 2);
    _Float16* WOUT = (_Float16*)wsalloc((size_t)NLAYERS * 256  * 512 * 2);
    _Float16* WXP  = (_Float16*)wsalloc((size_t)NLAYERS * 32   * 512 * 2);
    float*    HB   = (float*)   wsalloc((size_t)64 * 256 * 4);

    convert_weights_kernel<<<(NLAYERS * (1024*256 + 256*512 + 32*512) + 255) / 256, 256, 0, stream>>>(
        in_w, out_w, xp_w, WIN, WOUT, WXP);
    embed_kernel<<<(MROWS * 256 + 255) / 256, 256, 0, stream>>>(grid, ce, pe, X);

    for (int l = 0; l < NLAYERS; ++l) {
        // in-proj: XRH(f16) = X(f32) @ in_w + in_b     (M x 256) @ (256 x 1024)
        gemm_wmma<false, true><<<dim3(MROWS / 64, 1024 / 64), 128, 0, stream>>>(
            X, 256, WIN + (size_t)l * 1024 * 256, in_b + l * 1024,
            nullptr, 0, XRH, 1024, 1024, 256);
        // depthwise causal conv + SiLU (f16 -> f16)
        conv_silu_kernel<<<(MROWS * 512 + 255) / 256, 256, 0, stream>>>(
            XRH, conv_w + l * 512 * 4, conv_b + l * 512, XSCH);
        // x-proj: SSM(f32) = XSCH(f16) @ xproj_w + xproj_b   (M x 512) @ (512 x 32)
        gemm_wmma<true, false><<<dim3(MROWS / 64, 1), 128, 0, stream>>>(
            XSCH, 512, WXP + (size_t)l * 32 * 512, xp_b + l * 32,
            nullptr, 0, SSM, 32, 32, 512);
        // selective scan + fused gate (writes gated y, f16, in place over XSCH)
        scan_kernel<<<BATCH * (D_INNER / 16), 256, 0, stream>>>(
            SSM, A_log + l * 512 * 16, Dp + l * 512, XRH, XSCH);
        // out-proj with fused residual: X(f32) = X + y_gated(f16) @ out_w + out_b
        gemm_wmma<true, false><<<dim3(MROWS / 64, 256 / 64), 128, 0, stream>>>(
            XSCH, 512, WOUT + (size_t)l * 256 * 512, out_b + l * 256,
            X, 256, X, 256, 256, 512);
        // LayerNorm in place
        layernorm_kernel<<<MROWS, 256, 0, stream>>>(X, ln_g, ln_b);
    }

    heads1_kernel<<<64, 128, 0, stream>>>(X, c1w, c1b, l1w, l1b, HB);
    heads2_kernel<<<64, 128, 0, stream>>>(HB, c2w, c2b, l2w, l2b, outp);
}
